// BDA_24429773979892
// MI455X (gfx1250) — matrix operand order repaired and checked by reading, not compiled, not run on hardware
//
#include <hip/hip_runtime.h>
#include <hip/hip_bf16.h>
#include <math.h>

// ---------------------------------------------------------------------------
// Swin (2 blocks) + SE gate for MI455X (gfx1250, wave32, WMMA 16x16x32 f16)
// GEMM: async-to-LDS double-buffered pipeline, 16x64 wave tiles.
// ---------------------------------------------------------------------------

typedef __attribute__((ext_vector_type(16))) _Float16 v16h;
typedef __attribute__((ext_vector_type(8)))  _Float16 v8h;
typedef __attribute__((ext_vector_type(8)))  float    v8f;

#define B_   16
#define C_   256
#define H_   64
#define W_   64
#define WS_  8
#define NH_  8
#define HD_  32
#define N_   64          // tokens per window
#define L_   4096        // H*W
#define NW_  64          // windows per image
#define NTOK 65536       // B*L
#define SCALE_ 0.17677669529663687f   // 32^-0.5

// ---------------------------------------------------------------------------
// helpers
// ---------------------------------------------------------------------------
__device__ __forceinline__ float wave_reduce_add(float v) {
  #pragma unroll
  for (int m = 1; m < 32; m <<= 1) v += __shfl_xor(v, m, 32);
  return v;
}

__device__ __forceinline__ int shift_region(int p) {
  // H=W=64, WS=8, shift=4: slices [0,56),[56,60),[60,64)
  return (p < 56) ? 0 : ((p < 60) ? 1 : 2);
}

// gfx1250 async copy: 16B global -> LDS per lane, tracked by ASYNCcnt
__device__ __forceinline__ void async_copy_b128(void* lds, const void* gptr) {
  unsigned l = (unsigned)(uintptr_t)lds;                 // LDS byte address
  unsigned long long g = (unsigned long long)(uintptr_t)gptr;
  asm volatile("global_load_async_to_lds_b128 %0, %1, off"
               :: "v"(l), "v"(g) : "memory");
}
__device__ __forceinline__ void wait_async_le3() {
  asm volatile("s_wait_asynccnt 3" ::: "memory");
}
__device__ __forceinline__ void wait_async_le0() {
  asm volatile("s_wait_asynccnt 0" ::: "memory");
}

// ---------------------------------------------------------------------------
// weight transpose + f16 convert:  dst[n*K+k] = (f16)src[k*N+n]
// ---------------------------------------------------------------------------
__global__ void wtrans_kernel(const float* __restrict__ src,
                              _Float16* __restrict__ dst, int K, int N) {
  int idx = blockIdx.x * 256 + threadIdx.x;
  if (idx < K * N) {
    int k = idx / N, n = idx % N;
    dst[(size_t)n * K + k] = (_Float16)src[idx];
  }
}

// x [B,C,H,W] -> t [B,L,C] fp32
__global__ void x2t_kernel(const float* __restrict__ x, float* __restrict__ t) {
  int idx = blockIdx.x * 256 + threadIdx.x;   // over B*C*L = 2^24
  int b  = idx >> 20;
  int rem = idx & ((1 << 20) - 1);
  int c  = rem >> 12;
  int hw = rem & 4095;
  t[((size_t)(b << 12) + hw) * C_ + c] = x[idx];
}

// ---------------------------------------------------------------------------
// LayerNorm (+ optional window partition with cyclic shift), fp32 -> f16
// one wave per destination token
// ---------------------------------------------------------------------------
__global__ __launch_bounds__(256)
void ln_kernel(const float* __restrict__ t, const float* __restrict__ g,
               const float* __restrict__ bta, _Float16* __restrict__ out,
               int shift, int windowed) {
  int wave = threadIdx.x >> 5, lane = threadIdx.x & 31;
  int tok = blockIdx.x * 8 + wave;
  int srcTok;
  if (windowed) {
    int bb = tok >> 12, widx = (tok >> 6) & 63, n = tok & 63;
    int wr = widx >> 3, wc = widx & 7;
    int hh = ((wr << 3) + (n >> 3) + shift) & 63;
    int ww = ((wc << 3) + (n & 7) + shift) & 63;
    srcTok = (bb << 12) + (hh << 6) + ww;
  } else {
    srcTok = tok;
  }
  const float* xp = t + (size_t)srcTok * C_ + lane * 8;
  float4 x0 = *(const float4*)xp;
  float4 x1 = *(const float4*)(xp + 4);
  float xs[8] = {x0.x, x0.y, x0.z, x0.w, x1.x, x1.y, x1.z, x1.w};
  float s = 0.f;
  #pragma unroll
  for (int i = 0; i < 8; ++i) s += xs[i];
  s = wave_reduce_add(s);
  float mu = s * (1.0f / C_);
  float vs = 0.f;
  #pragma unroll
  for (int i = 0; i < 8; ++i) { float d = xs[i] - mu; vs += d * d; }
  vs = wave_reduce_add(vs);
  float inv = rsqrtf(vs * (1.0f / C_) + 1e-5f);
  _Float16* op = out + (size_t)tok * C_ + lane * 8;
  #pragma unroll
  for (int i = 0; i < 8; ++i) {
    int c = lane * 8 + i;
    op[i] = (_Float16)((xs[i] - mu) * inv * g[c] + bta[c]);
  }
}

// ---------------------------------------------------------------------------
// WMMA GEMM with async-LDS double buffering.
//  out = epilogue( A[M,K](f16) @ WT[N,K]^T(f16) + bias )
//  block = 256 thr (8 waves). Block tile 128(M) x 64(N); wave tile 16x64.
//  Per 32-wide K chunk: A(128x32)=8KB, B(64x32)=4KB staged via
//  GLOBAL_LOAD_ASYNC_TO_LDS_B128 (3 async ops per thread per chunk).
// modes: 0 f16 store | 1 f16*SCALE (Q) | 2 f16 V-transpose scatter
//        3 GELU f16 | 4 proj residual add (window reverse + unshift)
//        5 mlp2 residual add
// ---------------------------------------------------------------------------
__global__ __launch_bounds__(256)
void gemm_kernel(const _Float16* __restrict__ A, const _Float16* __restrict__ WT,
                 const float* __restrict__ bias, _Float16* __restrict__ outh,
                 float* __restrict__ resid, int M, int N, int K,
                 int mode, int shift) {
  __shared__ _Float16 Alds[2][128 * 32];
  __shared__ _Float16 Blds[2][64 * 32];

  int tid  = threadIdx.x;
  int wave = tid >> 5, lane = tid & 31;
  int hs  = lane >> 4;       // half-wave select
  int l15 = lane & 15;
  int rowBlk = blockIdx.x * 128;
  int colBlk = blockIdx.y * 64;

  // per-thread async-copy assignments (16B each)
  // A: 512 transfers (2/thread), B: 256 transfers (1/thread)
  int a0r = (tid)       >> 2, a0s = (tid)       & 3;
  int a1r = (tid + 256) >> 2, a1s = (tid + 256) & 3;
  int bcl = tid >> 2,         bsg = tid & 3;

  const _Float16* Abase = A  + (size_t)rowBlk * K;
  const _Float16* Bbase = WT + (size_t)colBlk * K;

  int nch = K >> 5;
  // prologue: stage chunk 0 into buffer 0
  {
    async_copy_b128(&Alds[0][a0r * 32 + a0s * 8], Abase + (size_t)a0r * K + a0s * 8);
    async_copy_b128(&Alds[0][a1r * 32 + a1s * 8], Abase + (size_t)a1r * K + a1s * 8);
    async_copy_b128(&Blds[0][bcl * 32 + bsg * 8], Bbase + (size_t)bcl * K + bsg * 8);
  }

  v8f acc[4] = {};
  int p = 0;
  for (int ch = 0; ch < nch; ++ch) {
    if (ch + 1 < nch) {
      int kk = (ch + 1) << 5;
      int q = p ^ 1;
      async_copy_b128(&Alds[q][a0r * 32 + a0s * 8], Abase + (size_t)a0r * K + kk + a0s * 8);
      async_copy_b128(&Alds[q][a1r * 32 + a1s * 8], Abase + (size_t)a1r * K + kk + a1s * 8);
      async_copy_b128(&Blds[q][bcl * 32 + bsg * 8], Bbase + (size_t)bcl * K + kk + bsg * 8);
      wait_async_le3();          // current chunk complete (in-order per wave)
    } else {
      wait_async_le0();
    }
    __syncthreads();

    const _Float16* Ap = &Alds[p][(wave * 16 + l15) * 32 + hs * 16];
    union { v16h v; v8h h[2]; } af;
    af.h[0] = *(const v8h*)(Ap);
    af.h[1] = *(const v8h*)(Ap + 8);
    #pragma unroll
    for (int ct = 0; ct < 4; ++ct) {
      const _Float16* Bp = &Blds[p][(ct * 16 + l15) * 32 + hs * 16];
      v16h bf = *(const v16h*)Bp;
      acc[ct] = __builtin_amdgcn_wmma_f32_16x16x32_f16(false, af.v, false, bf,
                                                       (short)0, acc[ct], false, false);
    }
    __syncthreads();             // reads done before this buffer is overwritten
    p ^= 1;
  }

  // epilogue
  #pragma unroll
  for (int ct = 0; ct < 4; ++ct) {
    int col = colBlk + ct * 16 + l15;
    float bv = bias[col];
    #pragma unroll
    for (int j = 0; j < 8; ++j) {
      int row = rowBlk + wave * 16 + hs * 8 + j;
      float v = acc[ct][j] + bv;
      if (mode == 0) {
        outh[(size_t)row * N + col] = (_Float16)v;
      } else if (mode == 1) {
        outh[(size_t)row * N + col] = (_Float16)(v * SCALE_);
      } else if (mode == 2) {
        int winIdx = row >> 6, n = row & 63;
        int head = col >> 5, hd = col & 31;
        outh[(((size_t)winIdx * NH_ + head) * HD_ + hd) * N_ + n] = (_Float16)v;
      } else if (mode == 3) {
        float ge = 0.5f * v * (1.0f + erff(v * 0.70710678118654752f));
        outh[(size_t)row * N + col] = (_Float16)ge;
      } else if (mode == 4) {
        int b  = row >> 12, widx = (row >> 6) & 63, n = row & 63;
        int wr = widx >> 3, wc = widx & 7;
        int hh = ((wr << 3) + (n >> 3) + shift) & 63;
        int ww = ((wc << 3) + (n & 7) + shift) & 63;
        size_t dst = ((size_t)(b << 12) + (hh << 6) + ww) * C_ + col;
        resid[dst] += v;
      } else { // mode 5
        resid[(size_t)row * C_ + col] += v;
      }
    }
  }
}

// ---------------------------------------------------------------------------
// Window attention: one block per (window, head); 128 thr (4 waves)
//  S = Q @ K^T (WMMA) -> LDS -> +rel-pos-bias (+shift mask) -> softmax
//  P(f16, LDS) @ V^T(global, pre-transposed) -> O (WMMA)
// ---------------------------------------------------------------------------
__global__ __launch_bounds__(128)
void attn_kernel(const _Float16* __restrict__ Q, const _Float16* __restrict__ Km,
                 const _Float16* __restrict__ Vt, const float* __restrict__ rpb,
                 _Float16* __restrict__ Out, int shift) {
  __shared__ float    S[N_][N_];
  __shared__ _Float16 P[N_][N_];

  int win  = blockIdx.x;        // 0..1023 (b*64+widx)
  int head = blockIdx.y;        // 0..7
  int wave = threadIdx.x >> 5, lane = threadIdx.x & 31;
  int hs = lane >> 4, l15 = lane & 15;
  size_t tokBase = (size_t)win * N_;

  // ---- scores: wave w handles rows 16w..16w+15, K=32 (one WMMA per tile)
  const _Float16* qrow = Q + (tokBase + wave * 16 + l15) * C_ + head * HD_ + hs * 16;
  union { v16h v; v8h h[2]; } af;
  af.h[0] = *(const v8h*)(qrow);
  af.h[1] = *(const v8h*)(qrow + 8);
  #pragma unroll
  for (int nt = 0; nt < 4; ++nt) {
    const _Float16* kp = Km + (tokBase + nt * 16 + l15) * C_ + head * HD_ + hs * 16;
    v16h bf = *(const v16h*)kp;
    v8f acc = {};
    acc = __builtin_amdgcn_wmma_f32_16x16x32_f16(false, af.v, false, bf,
                                                 (short)0, acc, false, false);
    #pragma unroll
    for (int j = 0; j < 8; ++j)
      S[wave * 16 + hs * 8 + j][nt * 16 + l15] = acc[j];
  }
  __syncthreads();

  // ---- softmax per row (threads 0..63, one row each)
  if (threadIdx.x < N_) {
    int r = threadIdx.x;
    int widx = win & 63, wr = widx >> 3, wc = widx & 7;
    int qh = r >> 3, qw = r & 7;
    int regq = 0;
    if (shift) regq = shift_region(wr * 8 + qh) * 3 + shift_region(wc * 8 + qw);
    float mx = -1e30f;
    for (int m = 0; m < N_; ++m) {
      int kh = m >> 3, kw = m & 7;
      int rel = (qh - kh + 7) * 15 + (qw - kw + 7);
      float v = S[r][m] + rpb[rel * NH_ + head];
      if (shift) {
        int regk = shift_region(wr * 8 + kh) * 3 + shift_region(wc * 8 + kw);
        if (regk != regq) v -= 100.0f;
      }
      S[r][m] = v;
      mx = fmaxf(mx, v);
    }
    float sum = 0.f;
    for (int m = 0; m < N_; ++m) {
      float e = __expf(S[r][m] - mx);
      S[r][m] = e;
      sum += e;
    }
    float inv = 1.0f / sum;
    for (int m = 0; m < N_; ++m) P[r][m] = (_Float16)(S[r][m] * inv);
  }
  __syncthreads();

  // ---- O = P @ V : wave w -> rows 16w..16w+15, cols 0..31, K=64
  v8f o0 = {}, o1 = {};
  #pragma unroll
  for (int kk = 0; kk < N_; kk += 32) {
    const _Float16* pr = &P[wave * 16 + l15][kk + hs * 16];
    union { v16h v; v8h h[2]; } pa;
    pa.h[0] = *(const v8h*)pr;
    pa.h[1] = *(const v8h*)(pr + 8);
    const _Float16* vb0 = Vt + (((size_t)win * NH_ + head) * HD_ + 0  + l15) * N_ + kk + hs * 16;
    const _Float16* vb1 = Vt + (((size_t)win * NH_ + head) * HD_ + 16 + l15) * N_ + kk + hs * 16;
    v16h b0 = *(const v16h*)vb0;
    v16h b1 = *(const v16h*)vb1;
    o0 = __builtin_amdgcn_wmma_f32_16x16x32_f16(false, pa.v, false, b0, (short)0, o0, false, false);
    o1 = __builtin_amdgcn_wmma_f32_16x16x32_f16(false, pa.v, false, b1, (short)0, o1, false, false);
  }
  #pragma unroll
  for (int j = 0; j < 8; ++j) {
    int tok = wave * 16 + hs * 8 + j;
    _Float16* op = Out + (tokBase + tok) * C_ + head * HD_;
    op[l15]      = (_Float16)o0[j];
    op[16 + l15] = (_Float16)o1[j];
  }
}

// ---------------------------------------------------------------------------
// SE tail
// ---------------------------------------------------------------------------
__global__ __launch_bounds__(256)
void pool_kernel(const float* __restrict__ x, const float* __restrict__ t,
                 float* __restrict__ pooled) {
  int wave = threadIdx.x >> 5, lane = threadIdx.x & 31;
  int idx = blockIdx.x * 8 + wave;   // b*256+c
  int b = idx >> 8, c = idx & 255;
  const float* xp = x + ((size_t)b * C_ + c) * L_;
  const float* tp = t + (size_t)b * L_ * C_ + c;
  float s = 0.f;
  for (int hw = lane; hw < L_; hw += 32) s += xp[hw] - tp[(size_t)hw * C_];
  s = wave_reduce_add(s);
  if (lane == 0) pooled[idx] = s * (1.0f / L_);
}

__global__ __launch_bounds__(256)
void gate_kernel(const float* __restrict__ pooled,
                 const float* __restrict__ w1, const float* __restrict__ b1,
                 const float* __restrict__ w2, const float* __restrict__ b2,
                 float* __restrict__ g) {
  __shared__ float G1[B_ * 512];
  for (int idx = threadIdx.x; idx < B_ * 512; idx += 256) {
    int b = idx >> 9, j = idx & 511;
    const float* pp = pooled + b * C_;
    const float* wp = w1 + (size_t)j * C_;
    float s = b1[j];
    for (int c = 0; c < C_; ++c) s += pp[c] * wp[c];
    G1[idx] = fmaxf(s, 0.f);
  }
  __syncthreads();
  for (int idx = threadIdx.x; idx < B_ * C_; idx += 256) {
    int b = idx >> 8, c = idx & 255;
    const float* wp = w2 + (size_t)c * 512;
    float s = b2[c];
    for (int j = 0; j < 512; ++j) s += G1[b * 512 + j] * wp[j];
    g[idx] = 1.0f / (1.0f + __expf(-s));
  }
}

__global__ void final_kernel(const float* __restrict__ x, const float* __restrict__ t,
                             const float* __restrict__ g, float* __restrict__ out) {
  int idx = blockIdx.x * 256 + threadIdx.x;   // over B*C*L
  int b  = idx >> 20;
  int rem = idx & ((1 << 20) - 1);
  int c  = rem >> 12;
  int hw = rem & 4095;
  float sw = t[((size_t)(b << 12) + hw) * C_ + c];
  out[idx] = sw + x[idx] * g[b * C_ + c];
}

// ---------------------------------------------------------------------------
// host launcher
// ---------------------------------------------------------------------------
extern "C" void kernel_launch(void* const* d_in, const int* in_sizes, int n_in,
                              void* d_out, int out_size, void* d_ws, size_t ws_size,
                              hipStream_t stream) {
  const float* x      = (const float*)d_in[0];
  const float* ln1_g  = (const float*)d_in[1];
  const float* ln1_b  = (const float*)d_in[2];
  const float* qkv_w  = (const float*)d_in[3];
  const float* qkv_b  = (const float*)d_in[4];
  const float* proj_w = (const float*)d_in[5];
  const float* proj_b = (const float*)d_in[6];
  const float* rpb    = (const float*)d_in[7];
  const float* ln2_g  = (const float*)d_in[8];
  const float* ln2_b  = (const float*)d_in[9];
  const float* mlp1_w = (const float*)d_in[10];
  const float* mlp1_b = (const float*)d_in[11];
  const float* mlp2_w = (const float*)d_in[12];
  const float* mlp2_b = (const float*)d_in[13];
  const float* se1_w  = (const float*)d_in[14];
  const float* se1_b  = (const float*)d_in[15];
  const float* se2_w  = (const float*)d_in[16];
  const float* se2_b  = (const float*)d_in[17];

  char* ws = (char*)d_ws;
  float*    t    = (float*)   (ws);                         // 64 MB fp32 residual tokens
  _Float16* A16  = (_Float16*)(ws + 67108864);              // 32 MB f16 (LN out / attn O)
  _Float16* Qb   = (_Float16*)(ws + 100663296);             // 32 MB
  _Float16* Kb   = (_Float16*)(ws + 134217728);             // 32 MB
  _Float16* Vt   = (_Float16*)(ws + 167772160);             // 32 MB (pre-transposed V)
  _Float16* M1   = Qb;                                      // 64 MB (reuses Q+K) MLP hidden
  char*     wsw  = ws + 201326592;
  _Float16* qkvT  = (_Float16*)(wsw);                       // 2 x 768x256
  _Float16* projT = (_Float16*)(wsw + 786432);              // 2 x 256x256
  _Float16* mlp1T = (_Float16*)(wsw + 786432 + 262144);     // 2 x 512x256
  _Float16* mlp2T = (_Float16*)(wsw + 786432 + 262144 + 524288); // 2 x 256x512
  float*    pooled = (float*)  (wsw + 2097152);
  float*    gbuf   = (float*)  (wsw + 2097152 + 16384);

  for (int d = 0; d < 2; ++d) {
    wtrans_kernel<<<(256 * 768 + 255) / 256, 256, 0, stream>>>(
        qkv_w + (size_t)d * 256 * 768, qkvT + (size_t)d * 768 * 256, 256, 768);
    wtrans_kernel<<<(256 * 256 + 255) / 256, 256, 0, stream>>>(
        proj_w + (size_t)d * 256 * 256, projT + (size_t)d * 256 * 256, 256, 256);
    wtrans_kernel<<<(256 * 512 + 255) / 256, 256, 0, stream>>>(
        mlp1_w + (size_t)d * 256 * 512, mlp1T + (size_t)d * 512 * 256, 256, 512);
    wtrans_kernel<<<(512 * 256 + 255) / 256, 256, 0, stream>>>(
        mlp2_w + (size_t)d * 512 * 256, mlp2T + (size_t)d * 256 * 512, 512, 256);
  }

  x2t_kernel<<<(B_ * C_ * L_) / 256, 256, 0, stream>>>(x, t);

  for (int d = 0; d < 2; ++d) {
    int shift = (d == 0) ? 0 : WS_ / 2;
    const _Float16* qkvT_d  = qkvT  + (size_t)d * 768 * 256;
    const _Float16* projT_d = projT + (size_t)d * 256 * 256;
    const _Float16* mlp1T_d = mlp1T + (size_t)d * 512 * 256;
    const _Float16* mlp2T_d = mlp2T + (size_t)d * 256 * 512;

    ln_kernel<<<NTOK / 8, 256, 0, stream>>>(t, ln1_g + d * C_, ln1_b + d * C_,
                                            A16, shift, 1);
    gemm_kernel<<<dim3(NTOK / 128, 4), 256, 0, stream>>>(
        A16, qkvT_d,             qkv_b + d * 768,       Qb, nullptr,
        NTOK, 256, 256, 1, 0);
    gemm_kernel<<<dim3(NTOK / 128, 4), 256, 0, stream>>>(
        A16, qkvT_d + 256 * 256, qkv_b + d * 768 + 256, Kb, nullptr,
        NTOK, 256, 256, 0, 0);
    gemm_kernel<<<dim3(NTOK / 128, 4), 256, 0, stream>>>(
        A16, qkvT_d + 512 * 256, qkv_b + d * 768 + 512, Vt, nullptr,
        NTOK, 256, 256, 2, 0);
    attn_kernel<<<dim3(B_ * NW_, NH_), 128, 0, stream>>>(
        Qb, Kb, Vt, rpb + (size_t)d * 225 * NH_, A16, shift);
    gemm_kernel<<<dim3(NTOK / 128, 4), 256, 0, stream>>>(
        A16, projT_d, proj_b + d * C_, nullptr, t,
        NTOK, 256, 256, 4, shift);
    ln_kernel<<<NTOK / 8, 256, 0, stream>>>(t, ln2_g + d * C_, ln2_b + d * C_,
                                            A16, 0, 0);
    gemm_kernel<<<dim3(NTOK / 128, 8), 256, 0, stream>>>(
        A16, mlp1T_d, mlp1_b + d * 512, M1, nullptr,
        NTOK, 512, 256, 3, 0);
    gemm_kernel<<<dim3(NTOK / 128, 4), 256, 0, stream>>>(
        M1, mlp2T_d, mlp2_b + d * C_, nullptr, t,
        NTOK, 256, 512, 5, 0);
  }

  pool_kernel<<<(B_ * C_) / 8, 256, 0, stream>>>(x, t, pooled);
  gate_kernel<<<1, 256, 0, stream>>>(pooled, se1_w, se1_b, se2_w, se2_b, gbuf);
  final_kernel<<<(B_ * C_ * L_) / 256, 256, 0, stream>>>(x, t, gbuf, (float*)d_out);
}
